// AttentionTransformerBlock_53738630807590
// MI455X (gfx1250) — matrix operand order repaired and checked by reading, not compiled, not run on hardware
//
#include <hip/hip_runtime.h>
#include <hip/hip_bf16.h>

typedef __attribute__((ext_vector_type(16))) _Float16 v16h;
typedef __attribute__((ext_vector_type(8)))  float    v8f;

#define HEADS 8
#define WS    8
#define DIM   128
#define HD    16          // head dim
#define BATCH 2
#define IMG   256
#define WN    32          // windows per side (256/8)
#define NPIX  (IMG*IMG)   // 65536 pixels per image
#define BH    (BATCH*HEADS)

// -------------------------------------------------------------------------
// Stage 1: AvgPool(8x8) -> LeakyReLU(0.01) -> 1x1 conv to offsets & scales
// off is divided by [wnw, wnh] = 32 as in the reference.
// -------------------------------------------------------------------------
__global__ __launch_bounds__(128)
void pool_off_scl_kernel(const float* __restrict__ x,
                         const float* __restrict__ off_w, const float* __restrict__ off_b,
                         const float* __restrict__ scl_w, const float* __restrict__ scl_b,
                         float* __restrict__ off_f, float* __restrict__ scl_f) {
  __shared__ float pooled[DIM];
  int bid = blockIdx.x;                       // b*WN*WN + wh*WN + ww
  int ww = bid & 31, wh = (bid >> 5) & 31, b = bid >> 10;
  int c = threadIdx.x;
  const float* src = x + ((size_t)(b*DIM + c))*NPIX + (size_t)(wh*8)*IMG + ww*8;
  float s = 0.f;
  #pragma unroll
  for (int i = 0; i < 8; ++i)
    #pragma unroll
    for (int j = 0; j < 8; ++j) s += src[i*IMG + j];
  s *= (1.f/64.f);
  pooled[c] = s > 0.f ? s : 0.01f*s;          // leaky_relu
  __syncthreads();
  if (threadIdx.x < 32) {
    int which = threadIdx.x >> 4;             // 0 = off, 1 = scl
    int oc    = threadIdx.x & 15;             // 0..15 = 2*HEADS channels
    const float* w = which ? scl_w : off_w;
    float d = which ? scl_b[oc] : off_b[oc];
    for (int cc = 0; cc < DIM; ++cc) d += w[oc*DIM + cc]*pooled[cc];
    int head = oc >> 1, comp = oc & 1;        // reshape (b,2H,..)->(bh,2,..)
    int idx = ((b*HEADS + head)*2 + comp)*(WN*WN) + wh*WN + ww;
    if (which) scl_f[idx] = d;
    else       off_f[idx] = d * (1.f/32.f);   // / [wnw, wnh] (both 32)
  }
}

// -------------------------------------------------------------------------
// Stage 2: QKV 1x1 conv as WMMA GEMM:  D[384, 65536] = W[384,128] x X[128,65536]
// 256-thread block = 8 waves covering a 128(M) x 16(N) slab.
// The 128x16 fp32 activation tile is staged ONCE per block into LDS with
// coalesced loads, converted to f16 and laid out in B-fragment order, so each
// wave's B fragment is two contiguous ds_load_b128 instead of 16 strided
// global dword loads (8x HBM traffic reduction vs per-wave fetch).
// -------------------------------------------------------------------------
__global__ __launch_bounds__(256)
void qkv_gemm_kernel(const float* __restrict__ x, const float* __restrict__ lms,
                     const float* __restrict__ qkv_w, const float* __restrict__ qkv_b,
                     _Float16* __restrict__ q_pan, _Float16* __restrict__ q_lms,
                     _Float16* __restrict__ kbuf,  _Float16* __restrict__ vbuf) {
  int ntile  = blockIdx.x;                    // 0..4095 pixel tiles
  int mblock = blockIdx.y;                    // 0..2 : 128-row slab (q / k / v)
  int z      = blockIdx.z;                    // b + 2*tensor
  int b = z & 1, tensor = z >> 1;             // tensor: 0 = pan(x), 1 = lms
  if (tensor == 1 && mblock != 0) return;     // lms: only Q needed
  const float* src = (tensor ? lms : x) + (size_t)b*DIM*NPIX;

  // B-fragment-ordered staging: Xs[(kk*2+hi)*16 + col][e], c = kk*32+hi*16+e
  __shared__ __align__(32) _Float16 Xs[DIM*16];
  int tid = threadIdx.x;
  for (int idx = tid; idx < DIM*16; idx += 256) {
    int c = idx >> 4, pix = idx & 15;         // adjacent tid -> adjacent pixel
    float v = src[(size_t)c*NPIX + ntile*16 + pix];
    Xs[((c >> 4)*16 + pix)*16 + (c & 15)] = (_Float16)v;
  }
  __syncthreads();

  int wave = tid >> 5, lane = tid & 31;
  int mtile = mblock*8 + wave;                // 0..23 output-channel tile
  int hi = (lane >= 16) ? 1 : 0;
  int col = lane & 15;
  v8f acc = {};
  #pragma unroll
  for (int kk = 0; kk < 4; ++kk) {
    v16h a;
    #pragma unroll
    for (int e = 0; e < 16; ++e) {
      int Ka = e + hi*16;                                 // A-frag K map
      a[e] = (_Float16)qkv_w[(mtile*16 + col)*DIM + kk*32 + Ka];
    }
    v16h bf = *(const v16h*)(Xs + ((kk*2 + hi)*16 + col)*16);
    acc = __builtin_amdgcn_wmma_f32_16x16x32_f16(false, a, false, bf,
                                                 (short)0, acc, false, false);
  }
  int part = mtile >> 3;                      // 0=q 1=k 2=v
  #pragma unroll
  for (int r = 0; r < 8; ++r) {
    int oc  = mtile*16 + r + hi*8;
    int pix = ntile*16 + col;
    float v = acc[r] + qkv_b[oc];
    int ch = oc & 127;
    int head = ch >> 4, d = ch & 15;
    size_t dst = ((size_t)((b*HEADS + head)*HD + d))*NPIX + pix;
    _Float16 hv = (_Float16)v;
    if (tensor == 0) {
      if (part == 0)      q_pan[dst] = hv;
      else if (part == 1) kbuf[dst]  = hv;
      else                vbuf[dst]  = hv;
    } else {
      q_lms[dst] = hv;
    }
  }
}

// -------------------------------------------------------------------------
// Stage 3: fused per-(window,head) attention.
// Block = one (b, wh, ww, head): 256 threads = 8 waves.
//   waves 0-3: lms queries (stripes 0-3), waves 4-7: pan queries.
// Phases: deform coords -> bilinear sample K/V to LDS -> QK^T WMMA ->
//         scale + rel-bias + softmax in LDS -> PV WMMA -> f16 O out.
// -------------------------------------------------------------------------
__global__ __launch_bounds__(256)
void attn_kernel(const _Float16* __restrict__ q_pan, const _Float16* __restrict__ q_lms,
                 const _Float16* __restrict__ kbuf,  const _Float16* __restrict__ vbuf,
                 const float* __restrict__ off_f, const float* __restrict__ scl_f,
                 const float* __restrict__ rel_table,
                 _Float16* __restrict__ o_lms, _Float16* __restrict__ o_pan) {
  __shared__ _Float16 ksel[64][HD];
  __shared__ _Float16 vsel[64][HD];
  __shared__ _Float16 qs[2][64][HD];          // [0]=lms, [1]=pan
  __shared__ float    Ssc[8][16][64];         // per-wave score rows
  __shared__ float    coordx[64], coordy[64];

  int bid = blockIdx.x;
  int head = bid & 7; int t1 = bid >> 3;
  int ww = t1 & 31; t1 >>= 5;
  int wh = t1 & 31; int b = t1 >> 5;
  int bh = b*HEADS + head;
  int tid = threadIdx.x;
  int wbase = wh*WN + ww;

  // ---- deformed sample coordinates per token ----
  if (tid < 64) {
    int ya = tid >> 3, xa = tid & 7;
    float offx = off_f[(bh*2 + 0)*(WN*WN) + wbase];
    float offy = off_f[(bh*2 + 1)*(WN*WN) + wbase];
    float sclx = scl_f[(bh*2 + 0)*(WN*WN) + wbase];
    float scly = scl_f[(bh*2 + 1)*(WN*WN) + wbase];
    float bx = -1.f + 2.f*(float)(ww*8 + xa)*(1.f/255.f);   // linspace(-1,1,256)
    float by = -1.f + 2.f*(float)(wh*8 + ya)*(1.f/255.f);
    float wcx = (float)(2*xa - 7)*(1.f/255.f);              // centered window coords
    float wcy = (float)(2*ya - 7)*(1.f/255.f);
    coordx[tid] = bx + wcx*sclx + offx;
    coordy[tid] = by + wcy*scly + offy;
  }
  // ---- load both Q windows ----
  for (int idx = tid; idx < 64*HD; idx += 256) {
    int t = idx >> 4, d = idx & 15;
    int y = wh*8 + (t >> 3), xx = ww*8 + (t & 7);
    size_t si = ((size_t)(bh*HD + d))*NPIX + (size_t)y*IMG + xx;
    qs[0][t][d] = q_lms[si];
    qs[1][t][d] = q_pan[si];
  }
  __syncthreads();

  // ---- bilinear grid-sample K and V (align_corners=True, zero pad) ----
  for (int idx = tid; idx < 2*64*HD; idx += 256) {
    int sel = (idx >= 64*HD) ? 1 : 0;
    int j = idx & (64*HD - 1);
    int t = j >> 4, d = j & 15;
    float gx = (coordx[t] + 1.f)*0.5f*255.f;
    float gy = (coordy[t] + 1.f)*0.5f*255.f;
    float x0f = floorf(gx), y0f = floorf(gy);
    int x0 = (int)x0f, y0 = (int)y0f;
    float fx = gx - x0f, fy = gy - y0f;
    const _Float16* src = (sel ? vbuf : kbuf) + ((size_t)(bh*HD + d))*NPIX;
    float w00 = (1.f-fx)*(1.f-fy), w01 = fx*(1.f-fy);
    float w10 = (1.f-fx)*fy,       w11 = fx*fy;
    float acc = 0.f;
    if ((unsigned)x0     < IMG && (unsigned)y0     < IMG) acc += w00*(float)src[(size_t)y0*IMG + x0];
    if ((unsigned)(x0+1) < IMG && (unsigned)y0     < IMG) acc += w01*(float)src[(size_t)y0*IMG + x0 + 1];
    if ((unsigned)x0     < IMG && (unsigned)(y0+1) < IMG) acc += w10*(float)src[(size_t)(y0+1)*IMG + x0];
    if ((unsigned)(x0+1) < IMG && (unsigned)(y0+1) < IMG) acc += w11*(float)src[(size_t)(y0+1)*IMG + x0 + 1];
    if (sel) vsel[t][d] = (_Float16)acc; else ksel[t][d] = (_Float16)acc;
  }
  __syncthreads();

  int wave = tid >> 5, lane = tid & 31;
  int selq   = wave >> 2;                     // 0=lms, 1=pan
  int stripe = wave & 3;                      // 16-query stripe
  int hi = (lane >= 16) ? 1 : 0;
  int col = lane & 15;

  // ---- S = Q K^T : hd=16 zero-padded to K=32, one WMMA per 16-key tile ----
  v16h aq;
  #pragma unroll
  for (int e = 0; e < 16; ++e) {
    int K = e + hi*16;
    aq[e] = (K < HD) ? qs[selq][stripe*16 + col][K] : (_Float16)0.f;
  }
  #pragma unroll
  for (int kt = 0; kt < 4; ++kt) {
    v16h bk;
    #pragma unroll
    for (int e = 0; e < 16; ++e) {
      int K = e + hi*16;
      bk[e] = (K < HD) ? ksel[kt*16 + col][K] : (_Float16)0.f;
    }
    v8f sc = {};
    sc = __builtin_amdgcn_wmma_f32_16x16x32_f16(false, aq, false, bk,
                                                (short)0, sc, false, false);
    #pragma unroll
    for (int r = 0; r < 8; ++r)
      Ssc[wave][r + hi*8][kt*16 + col] = sc[r];
  }
  __syncthreads();

  // ---- scale + relative-position bias + softmax (one query row per lane) ----
  if (lane < 16) {
    int qtok = stripe*16 + lane;
    int qy = qtok >> 3, qx = qtok & 7;
    float mx = -1e30f;
    for (int k = 0; k < 64; ++k) {
      int ky = k >> 3, kx = k & 7;
      int ridx = (qy - ky + 7)*15 + (qx - kx + 7);
      float s = Ssc[wave][lane][k]*0.25f + rel_table[ridx*HEADS + head];
      Ssc[wave][lane][k] = s;
      mx = fmaxf(mx, s);
    }
    float sum = 0.f;
    for (int k = 0; k < 64; ++k) {
      float e = expf(Ssc[wave][lane][k] - mx);
      Ssc[wave][lane][k] = e;
      sum += e;
    }
    float inv = 1.f/sum;
    for (int k = 0; k < 64; ++k) Ssc[wave][lane][k] *= inv;
  }
  __syncthreads();

  // ---- O = P V : K=64 keys as two K=32 WMMA steps ----
  v8f oacc = {};
  #pragma unroll
  for (int ch = 0; ch < 2; ++ch) {
    v16h ap, bv;
    #pragma unroll
    for (int e = 0; e < 16; ++e) {
      int K  = ch*32 + e + hi*16;                           // key index (A)
      ap[e] = (_Float16)Ssc[wave][col][K];
      int Kb = ch*32 + e + hi*16;                           // key index (B)
      bv[e] = vsel[Kb][col];
    }
    oacc = __builtin_amdgcn_wmma_f32_16x16x32_f16(false, ap, false, bv,
                                                  (short)0, oacc, false, false);
  }
  _Float16* dst = selq ? o_pan : o_lms;
  #pragma unroll
  for (int r = 0; r < 8; ++r) {
    int qtok = stripe*16 + r + hi*8;
    int y = wh*8 + (qtok >> 3), xx = ww*8 + (qtok & 7);
    dst[((size_t)(b*DIM + head*HD + col))*NPIX + (size_t)y*IMG + xx] = (_Float16)oacc[r];
  }
}

// -------------------------------------------------------------------------
// Stage 4: projection GEMM: out[128,N] = proj_w[128,128] x O[128,N] + b.
// One 8-wave block covers all 128 output rows for a 16-pixel tile; the
// 128x16 f16 O tile is staged once into LDS in B-fragment order.
// -------------------------------------------------------------------------
__global__ __launch_bounds__(256)
void proj_gemm_kernel(const _Float16* __restrict__ o_lms, const _Float16* __restrict__ o_pan,
                      const float* __restrict__ proj_w, const float* __restrict__ proj_b,
                      float* __restrict__ out) {
  int ntile = blockIdx.x;                     // 0..4095
  int z = blockIdx.y;                         // b + 2*sel
  int b = z & 1, sel = z >> 1;                // sel: 0=lms, 1=pan
  const _Float16* src = (sel ? o_pan : o_lms) + (size_t)b*DIM*NPIX;

  __shared__ __align__(32) _Float16 Os[DIM*16];
  int tid = threadIdx.x;
  for (int idx = tid; idx < DIM*16; idx += 256) {
    int c = idx >> 4, pix = idx & 15;
    Os[((c >> 4)*16 + pix)*16 + (c & 15)] = src[(size_t)c*NPIX + ntile*16 + pix];
  }
  __syncthreads();

  int wave = tid >> 5, lane = tid & 31;       // wave = mtile 0..7
  int hi = (lane >= 16) ? 1 : 0;
  int col = lane & 15;
  v8f acc = {};
  #pragma unroll
  for (int kk = 0; kk < 4; ++kk) {
    v16h a;
    #pragma unroll
    for (int e = 0; e < 16; ++e) {
      int Ka = e + hi*16;
      a[e] = (_Float16)proj_w[(wave*16 + col)*DIM + kk*32 + Ka];
    }
    v16h bf = *(const v16h*)(Os + ((kk*2 + hi)*16 + col)*16);
    acc = __builtin_amdgcn_wmma_f32_16x16x32_f16(false, a, false, bf,
                                                 (short)0, acc, false, false);
  }
  float* dstbase = out + (size_t)sel*((size_t)BATCH*DIM*NPIX);
  #pragma unroll
  for (int r = 0; r < 8; ++r) {
    int oc = wave*16 + r + hi*8;
    dstbase[((size_t)(b*DIM + oc))*NPIX + ntile*16 + col] = acc[r] + proj_b[oc];
  }
}

// -------------------------------------------------------------------------
extern "C" void kernel_launch(void* const* d_in, const int* in_sizes, int n_in,
                              void* d_out, int out_size, void* d_ws, size_t ws_size,
                              hipStream_t stream) {
  (void)in_sizes; (void)n_in; (void)out_size; (void)ws_size;
  const float* x         = (const float*)d_in[0];
  const float* lms       = (const float*)d_in[1];
  const float* qkv_w     = (const float*)d_in[2];
  const float* qkv_b     = (const float*)d_in[3];
  const float* proj_w    = (const float*)d_in[4];
  const float* proj_b    = (const float*)d_in[5];
  const float* off_w     = (const float*)d_in[6];
  const float* off_b     = (const float*)d_in[7];
  const float* scl_w     = (const float*)d_in[8];
  const float* scl_b     = (const float*)d_in[9];
  const float* rel_table = (const float*)d_in[10];
  float* out = (float*)d_out;

  // carve workspace (all intermediates f16 except off/scl)
  char* p = (char*)d_ws;
  auto alloc = [&](size_t bytes) -> void* {
    void* r = (void*)p;
    p += (bytes + 255) & ~(size_t)255;
    return r;
  };
  float*    off_f = (float*)   alloc((size_t)BH*2*WN*WN*sizeof(float));
  float*    scl_f = (float*)   alloc((size_t)BH*2*WN*WN*sizeof(float));
  _Float16* q_pan = (_Float16*)alloc((size_t)BH*HD*NPIX*sizeof(_Float16));
  _Float16* q_lms = (_Float16*)alloc((size_t)BH*HD*NPIX*sizeof(_Float16));
  _Float16* kbuf  = (_Float16*)alloc((size_t)BH*HD*NPIX*sizeof(_Float16));
  _Float16* vbuf  = (_Float16*)alloc((size_t)BH*HD*NPIX*sizeof(_Float16));
  _Float16* olms  = (_Float16*)alloc((size_t)BATCH*DIM*NPIX*sizeof(_Float16));
  _Float16* opan  = (_Float16*)alloc((size_t)BATCH*DIM*NPIX*sizeof(_Float16));

  // 1) pooled offsets / scales
  pool_off_scl_kernel<<<BATCH*WN*WN, 128, 0, stream>>>(
      x, off_w, off_b, scl_w, scl_b, off_f, scl_f);

  // 2) QKV GEMM (WMMA f16, LDS-staged B): 4096 x 3 x {b,tensor}
  qkv_gemm_kernel<<<dim3(NPIX/16, 3, 4), 256, 0, stream>>>(
      x, lms, qkv_w, qkv_b, q_pan, q_lms, kbuf, vbuf);

  // 3) fused deformable window attention: one block per (b, wh, ww, head)
  attn_kernel<<<BATCH*WN*WN*HEADS, 256, 0, stream>>>(
      q_pan, q_lms, kbuf, vbuf, off_f, scl_f, rel_table, olms, opan);

  // 4) projection GEMM -> d_out (lms output first, then pan)
  proj_gemm_kernel<<<dim3(NPIX/16, 4), 256, 0, stream>>>(
      olms, opan, proj_w, proj_b, out);
}
